// MinHashSketch_69793218560122
// MI455X (gfx1250) — matrix operand order repaired and checked by reading, not compiled, not run on hardware
//
#include <hip/hip_runtime.h>
#include <hip/hip_bf16.h>

// CDNA5 / gfx1250, wave32. WMMA bf16 path with split-precision (hi+lo) fp32 emulation.

typedef __attribute__((ext_vector_type(16))) __bf16 v16bf;
typedef __attribute__((ext_vector_type(8)))  float  v8f;

#define FEAT        256   // FEATURE_DIM
#define NHASH       128   // NUM_HASHES
#define BLOCK_ROWS  128   // nodes per workgroup (8 waves x 16-row M tiles)
#define N_TILES     8     // 128 / 16
#define K_ITERS     8     // 256 / 32
#define LDS_STRIDE  132   // 128 + 4 pad: avoids bank conflicts between lane halves

__device__ __forceinline__ unsigned short f2bf_rne(float f) {
  unsigned u = __float_as_uint(f);
  unsigned r = u + 0x7FFFu + ((u >> 16) & 1u);   // round-to-nearest-even to bf16
  return (unsigned short)(r >> 16);
}
__device__ __forceinline__ float bf2f(unsigned short s) {
  return __uint_as_float(((unsigned)s) << 16);
}
// order-preserving map f32 -> u32 so unsigned atomicMin == float min
__device__ __forceinline__ unsigned f32_key(float f) {
  unsigned u = __float_as_uint(f);
  return (u & 0x80000000u) ? ~u : (u | 0x80000000u);
}

union V16 { v16bf v; unsigned short u[16]; };

// ---------------------------------------------------------------------------
// Kernel 1: swizzle hash matrix into per-lane WMMA B-fragment layout (bf16 hi/lo)
// and (re)initialize the min-key buffer. Runs every call: deterministic.
// B frag layout (32x16 KxN bf16, wave32): lane = 16*half + n; lane holds
// K = kiter*32 + 16*half + [0..15] contiguous -> 32B per lane, coalesced.
// element index i = ((ntile*8 + kiter)*32 + lane)*16 + j
// ---------------------------------------------------------------------------
__global__ __launch_bounds__(256) void minhash_prep(
    const float* __restrict__ hm, unsigned short* __restrict__ Bhi,
    unsigned short* __restrict__ Blo, unsigned* __restrict__ keys, int nKeys) {
  int i = blockIdx.x * 256 + threadIdx.x;
  if (i < N_TILES * K_ITERS * 32 * 16) {
    int j     = i & 15;
    int lane  = (i >> 4) & 31;
    int kiter = (i >> 9) & 7;
    int ntile = i >> 12;
    int n = ntile * 16 + (lane & 15);
    int k = kiter * 32 + 16 * (lane >> 4) + j;
    float v = hm[n * FEAT + k];
    unsigned short hi = f2bf_rne(v);
    unsigned short lo = f2bf_rne(v - bf2f(hi));
    Bhi[i] = hi;
    Blo[i] = lo;
  }
  if (i < nKeys) keys[i] = 0xFFFFFFFFu;
}

// ---------------------------------------------------------------------------
// Kernel 2: fused GEMM (split-bf16 WMMA) + sorted-segment min reduction.
// Block = 256 threads = 8 waves; wave w owns rows [base + 16w, +16), all 128 cols.
// ---------------------------------------------------------------------------
__global__ __launch_bounds__(256) void minhash_gemm_min(
    const float* __restrict__ x, const long long* __restrict__ batch,
    const unsigned short* __restrict__ Bhi, const unsigned short* __restrict__ Blo,
    unsigned* __restrict__ keys, int numNodes) {
  __shared__ float ldsC[BLOCK_ROWS * LDS_STRIDE];
  __shared__ int   ldsSeg[BLOCK_ROWS];

  const int t    = threadIdx.x;
  const int lane = t & 31;
  const int wave = t >> 5;                 // 0..7
  const int base = blockIdx.x * BLOCK_ROWS;

  if (t < BLOCK_ROWS) {
    int node = base + t;
    ldsSeg[t] = (node < numNodes) ? (int)batch[node] : -1;
  }

  const int mrow = lane & 15;
  const int half = lane >> 4;              // 0/1: K/M interleave half
  int node = base + wave * 16 + mrow;
  if (node >= numNodes) node = numNodes - 1;   // clamp: garbage rows never reduced
  const float* xrow = x + (size_t)node * FEAT;

  v8f c[N_TILES];
#pragma unroll
  for (int n = 0; n < N_TILES; ++n)
#pragma unroll
    for (int i = 0; i < 8; ++i) c[n][i] = 0.0f;

  for (int k = 0; k < K_ITERS; ++k) {
    // A (16x32 bf16 MxK) per-lane K set: half=0 -> {0..15}, half=1 -> {16..31}
    const float* p = xrow + k * 32 + 16 * half;
    float4 q0 = *(const float4*)(p);
    float4 q1 = *(const float4*)(p + 4);
    float4 q2 = *(const float4*)(p + 8);
    float4 q3 = *(const float4*)(p + 12);
    float a[16] = {q0.x, q0.y, q0.z, q0.w, q1.x, q1.y, q1.z, q1.w,
                   q2.x, q2.y, q2.z, q2.w, q3.x, q3.y, q3.z, q3.w};
    V16 ahi, alo;
#pragma unroll
    for (int i = 0; i < 16; ++i) {
      unsigned short h = f2bf_rne(a[i]);
      ahi.u[i] = h;
      alo.u[i] = f2bf_rne(a[i] - bf2f(h));
    }
#pragma unroll
    for (int n = 0; n < N_TILES; ++n) {
      int fi = ((n * K_ITERS + k) * 32 + lane) * 16;
      v16bf bh = *(const v16bf*)(Bhi + fi);
      v16bf bl = *(const v16bf*)(Blo + fi);
      // x*y ~= xh*yh + xh*yl + xl*yh  (fp32-accurate, all in one f32 accumulator)
      c[n] = __builtin_amdgcn_wmma_f32_16x16x32_bf16(false, ahi.v, false, bh,
                                                     (short)0, c[n], false, false);
      c[n] = __builtin_amdgcn_wmma_f32_16x16x32_bf16(false, ahi.v, false, bl,
                                                     (short)0, c[n], false, false);
      c[n] = __builtin_amdgcn_wmma_f32_16x16x32_bf16(false, alo.v, false, bh,
                                                     (short)0, c[n], false, false);
    }
  }

  // C layout: VGPR r -> (M = r + 8*half, N = lane&15) within 16x16 tile
#pragma unroll
  for (int n = 0; n < N_TILES; ++n)
#pragma unroll
    for (int r = 0; r < 8; ++r)
      ldsC[(wave * 16 + 8 * half + r) * LDS_STRIDE + n * 16 + mrow] = c[n][r];

  __syncthreads();

  // Per-column run-length segment min over this block's rows; one atomic per run.
  const int col = t & 127;
  const int r0  = (t >> 7) * 64;
  int curSeg = -1;
  float curMin = 0.0f;
  for (int r = r0; r < r0 + 64; ++r) {
    if (base + r >= numNodes) break;
    int seg = ldsSeg[r];
    float v = ldsC[r * LDS_STRIDE + col];
    if (seg != curSeg) {
      if (curSeg >= 0)
        atomicMin(&keys[(size_t)curSeg * NHASH + col], f32_key(curMin));
      curSeg = seg;
      curMin = v;
    } else {
      curMin = fminf(curMin, v);
    }
  }
  if (curSeg >= 0)
    atomicMin(&keys[(size_t)curSeg * NHASH + col], f32_key(curMin));
}

// ---------------------------------------------------------------------------
// Kernel 3: decode keys -> floats; untouched (empty segment) -> 0 (inf -> 0 rule)
// ---------------------------------------------------------------------------
__global__ __launch_bounds__(256) void minhash_finalize(
    const unsigned* __restrict__ keys, float* __restrict__ out, int n) {
  int i = blockIdx.x * 256 + threadIdx.x;
  if (i >= n) return;
  unsigned k = keys[i];
  float v;
  if (k == 0xFFFFFFFFu) {
    v = 0.0f;
  } else {
    unsigned u = (k & 0x80000000u) ? (k ^ 0x80000000u) : ~k;
    v = __uint_as_float(u);
  }
  out[i] = v;
}

extern "C" void kernel_launch(void* const* d_in, const int* in_sizes, int n_in,
                              void* d_out, int out_size, void* d_ws, size_t ws_size,
                              hipStream_t stream) {
  const float*     x     = (const float*)d_in[0];
  const long long* batch = (const long long*)d_in[1];   // int64 segment ids (sorted)
  const float*     hm    = (const float*)d_in[3];       // [NHASH, FEAT] f32

  const int numNodes = in_sizes[1];
  const int numSeg   = out_size / NHASH;                // 512
  const int nKeys    = numSeg * NHASH;

  // workspace layout: [0,64K) B_hi bf16 | [64K,128K) B_lo bf16 | [128K, +256K) keys
  char* ws = (char*)d_ws;
  unsigned short* Bhi  = (unsigned short*)(ws);
  unsigned short* Blo  = (unsigned short*)(ws + 64 * 1024);
  unsigned*       keys = (unsigned*)(ws + 128 * 1024);

  int prepTotal = (N_TILES * K_ITERS * 32 * 16) > nKeys ? (N_TILES * K_ITERS * 32 * 16) : nKeys;
  minhash_prep<<<(prepTotal + 255) / 256, 256, 0, stream>>>(hm, Bhi, Blo, keys, nKeys);

  int nBlocks = (numNodes + BLOCK_ROWS - 1) / BLOCK_ROWS;
  minhash_gemm_min<<<nBlocks, 256, 0, stream>>>(x, batch, Bhi, Blo, keys, numNodes);

  minhash_finalize<<<(out_size + 255) / 256, 256, 0, stream>>>(keys, (float*)d_out, out_size);
}
